// RPNHeadWarper_76459007803565
// MI455X (gfx1250) — matrix operand order repaired and checked by reading, not compiled, not run on hardware
//
#include <hip/hip_runtime.h>

// ---------------------------------------------------------------------------
// RPN head post-processing for MI455X (gfx1250, wave32).
// Stages: per-level streaming top-2000 -> delta2bbox decode -> greedy NMS
// (boxes staged into LDS with the Tensor Data Mover, keep-rank prefix scan
// done with v_wmma_f32_16x16x32_f16 triangular matmuls) -> exact 5-way merge.
// ---------------------------------------------------------------------------

#define NMS_PRE  2000
#define NMS_POST 1000
#define MAX_NUM  1000
#define IMG_W    1344.0f
#define IMG_H    800.0f
#define MAX_RATIO 4.135166556742356f   // |log(16/1000)|

typedef _Float16      v16h  __attribute__((ext_vector_type(16)));
typedef float         v8f   __attribute__((ext_vector_type(8)));
typedef unsigned int  u32x4 __attribute__((ext_vector_type(4)));
typedef int           i32x4 __attribute__((ext_vector_type(4)));
typedef int           i32x8 __attribute__((ext_vector_type(8)));

// Total order matching lax.top_k / stable argsort(-s): value desc, index asc.
__device__ __forceinline__ bool beats(float sa, int ia, float sb, int ib) {
  return (sa > sb) || ((sa == sb) && (ia < ib));
}

// ---------------------------------------------------------------------------
// Stage 1: streaming bitonic top-2048 (keeps best 2048, we emit first 2000)
// ---------------------------------------------------------------------------
#define TK_N 2048
#define TK_T 512

__device__ void bitonic_sort_full(float* s, int* ix) {
  const int tid = threadIdx.x;
  for (int k = 2; k <= TK_N; k <<= 1) {
    for (int j = k >> 1; j > 0; j >>= 1) {
      for (int i = tid; i < TK_N; i += TK_T) {
        int p = i ^ j;
        if (p > i) {
          float s1 = s[i], s2 = s[p];
          int   i1 = ix[i], i2 = ix[p];
          bool up = ((i & k) == 0);
          bool sw = up ? beats(s2, i2, s1, i1) : beats(s1, i1, s2, i2);
          if (sw) { s[i] = s2; ix[i] = i2; s[p] = s1; ix[p] = i1; }
        }
      }
      __syncthreads();
    }
  }
}

__device__ void bitonic_merge_clean(float* s, int* ix) {
  const int tid = threadIdx.x;
  for (int j = TK_N >> 1; j > 0; j >>= 1) {
    for (int i = tid; i < TK_N; i += TK_T) {
      int p = i ^ j;
      if (p > i) {
        float s1 = s[i], s2 = s[p];
        int   i1 = ix[i], i2 = ix[p];
        if (beats(s2, i2, s1, i1)) { s[i] = s2; ix[i] = i2; s[p] = s1; ix[p] = i1; }
      }
    }
    __syncthreads();
  }
}

__global__ __launch_bounds__(TK_T)
void topk_kernel(const float* c0, const float* c1, const float* c2,
                 const float* c3, const float* c4,
                 float* __restrict__ outS, int* __restrict__ outI) {
  const int bl  = blockIdx.x;           // b*5 + lvl
  const int b   = bl / 5, lvl = bl % 5;
  const int HWs[5] = {67200, 16800, 4200, 1050, 273};
  const int HW    = HWs[lvl];
  const int Nreal = HW * 3;
  const int M     = (Nreal > NMS_PRE) ? Nreal : NMS_PRE;   // pad small level to 2000
  const float* cls;
  switch (lvl) { case 0: cls = c0; break; case 1: cls = c1; break;
                 case 2: cls = c2; break; case 3: cls = c3; break; default: cls = c4; }
  cls += (size_t)b * 3 * HW;

  __shared__ float bs[TK_N]; __shared__ int bi[TK_N];
  __shared__ float cs[TK_N]; __shared__ int ci[TK_N];
  const int tid = threadIdx.x;
  const int nch = (M + TK_N - 1) / TK_N;

  for (int ch = 0; ch < nch; ++ch) {
    float* s  = (ch == 0) ? bs : cs;
    int*   ix = (ch == 0) ? bi : ci;
    for (int t = tid; t < TK_N; t += TK_T) {
      int n = ch * TK_N + t;
      float sv; int iv;
      if (n < Nreal) {                       // flat order (H,W,A): a = n%3
        int a = n % 3, hw = n / 3;
        float x = cls[(size_t)a * HW + hw];
        sv = 1.0f / (1.0f + expf(-x));
        iv = n;
      } else if (n < M) { sv = -1.0f; iv = n; }          // reference -1 padding
      else { sv = -3.4e38f; iv = 0x7fffffff; }           // chunk sentinel
      s[t] = sv; ix[t] = iv;
    }
    __syncthreads();
    bitonic_sort_full(s, ix);                // best-first
    if (ch > 0) {
      // elementwise merge: buf (best-first) vs reversed chunk -> bitonic top-2048
      for (int i = tid; i < TK_N; i += TK_T) {
        int j = TK_N - 1 - i;
        float s2 = cs[j]; int i2 = ci[j];
        if (beats(s2, i2, bs[i], bi[i])) { bs[i] = s2; bi[i] = i2; }
      }
      __syncthreads();
      bitonic_merge_clean(bs, bi);
    }
  }
  for (int i = tid; i < NMS_PRE; i += TK_T) {
    outS[(size_t)bl * NMS_PRE + i] = bs[i];
    outI[(size_t)bl * NMS_PRE + i] = bi[i];
  }
}

// ---------------------------------------------------------------------------
// Stage 2: delta2bbox decode of the gathered top-2000
// ---------------------------------------------------------------------------
__global__ void decode_kernel(const float* b0, const float* b1, const float* b2,
                              const float* b3, const float* b4,
                              const float* a0, const float* a1, const float* a2,
                              const float* a3, const float* a4,
                              const int* __restrict__ topI,
                              float4* __restrict__ props) {
  int t = blockIdx.x * blockDim.x + threadIdx.x;
  if (t >= 2 * 5 * NMS_PRE) return;
  int b   = t / (5 * NMS_PRE);
  int rem = t % (5 * NMS_PRE);
  int lvl = rem / NMS_PRE, i = rem % NMS_PRE;
  int bl  = b * 5 + lvl;
  const int HWs[5] = {67200, 16800, 4200, 1050, 273};
  int HW = HWs[lvl], Nreal = HW * 3;
  const float* bbox; const float* anc;
  switch (lvl) { case 0: bbox = b0; anc = a0; break; case 1: bbox = b1; anc = a1; break;
                 case 2: bbox = b2; anc = a2; break; case 3: bbox = b3; anc = a3; break;
                 default: bbox = b4; anc = a4; }
  int idx = topI[(size_t)bl * NMS_PRE + i];
  float4 o = make_float4(0.f, 0.f, 0.f, 0.f);       // zero pads decode to (0,0,0,0)
  if (idx < Nreal) {
    int a = idx % 3, hw = idx / 3;
    const float* dp = bbox + ((size_t)b * 12 + (size_t)a * 4) * HW + hw;
    float dx = dp[0], dy = dp[HW];
    float dw = dp[2 * (size_t)HW], dh = dp[3 * (size_t)HW];
    dw = fminf(fmaxf(dw, -MAX_RATIO), MAX_RATIO);
    dh = fminf(fmaxf(dh, -MAX_RATIO), MAX_RATIO);
    float ax1 = anc[4 * (size_t)idx + 0], ay1 = anc[4 * (size_t)idx + 1];
    float ax2 = anc[4 * (size_t)idx + 2], ay2 = anc[4 * (size_t)idx + 3];
    float pw = ax2 - ax1, ph = ay2 - ay1;
    float px = (ax1 + ax2) * 0.5f, py = (ay1 + ay2) * 0.5f;
    float gw = pw * expf(dw), gh = ph * expf(dh);
    float gx = px + pw * dx,  gy = py + ph * dy;
    o.x = fminf(fmaxf(gx - 0.5f * gw, 0.f), IMG_W);
    o.y = fminf(fmaxf(gy - 0.5f * gh, 0.f), IMG_H);
    o.z = fminf(fmaxf(gx + 0.5f * gw, 0.f), IMG_W);
    o.w = fminf(fmaxf(gy + 0.5f * gh, 0.f), IMG_H);
  }
  props[(size_t)bl * NMS_PRE + i] = o;
}

// ---------------------------------------------------------------------------
// TDM: 1-D tile (ndwords x 1) Global -> LDS. Descriptor per ISA cdna5 §8.
// ---------------------------------------------------------------------------
__device__ __forceinline__ void tdm_load_tile(void* lds_dst, const void* gsrc, int bytes) {
  unsigned long long ga = (unsigned long long)gsrc;
  unsigned int lo = (unsigned int)((unsigned long long)lds_dst);  // LDS byte offset
  unsigned int ndw = (unsigned int)(bytes >> 2);
  u32x4 g0;
  g0.x = 1u;                                            // count=1, user descriptor
  g0.y = lo;                                            // lds_addr
  g0.z = (unsigned int)(ga & 0xFFFFFFFFull);            // global_addr[31:0]
  g0.w = (unsigned int)((ga >> 32) & 0x01FFFFFFull) | 0x80000000u; // addr[56:32] | type=2
  i32x8 g1;
  g1[0] = 0x00020000;                  // data_size=4B, no mask/iterate/pad/barrier
  g1[1] = (int)(ndw << 16);            // tensor_dim0[15:0]  (bits 63:48)
  g1[2] = (int)(((ndw >> 16) & 0xFFFFu) | (1u << 16)); // dim0 hi | tensor_dim1=1
  g1[3] = (int)(ndw << 16);            // tile_dim0 (bits 127:112)
  g1[4] = 1;                           // tile_dim1=1, tile_dim2=0
  g1[5] = (int)ndw;                    // tensor_dim0_stride lo
  g1[6] = 0; g1[7] = 0;
  i32x4 z4 = {0, 0, 0, 0};
#if defined(__clang_major__) && (__clang_major__ >= 23)
  i32x8 z8 = {0, 0, 0, 0, 0, 0, 0, 0};
  __builtin_amdgcn_tensor_load_to_lds(g0, g1, z4, z4, z8, 0);
#else
  __builtin_amdgcn_tensor_load_to_lds(g0, g1, z4, z4, 0);
#endif
}

// ---------------------------------------------------------------------------
// Stage 3: greedy NMS in LDS + WMMA (16x16x32 f16) prefix scan of keep flags
// ---------------------------------------------------------------------------
#define NMS_T 256

__global__ __launch_bounds__(NMS_T)
void nms_kernel(const float4* __restrict__ props, const float* __restrict__ topS,
                float* __restrict__ lvlS, float4* __restrict__ lvlB) {
  const int bl  = blockIdx.x;   // b*5 + lvl
  const int tid = threadIdx.x;
  __shared__ __align__(16) float4       box[NMS_PRE];
  __shared__ float                      sc[NMS_PRE];
  __shared__ unsigned char              supp[NMS_PRE];
  __shared__ __align__(32) _Float16     kf[2048];
  __shared__ int                        scanv[2048];
  __shared__ float                      ctot[8];

  // --- TDM: async DMA of the 32KB proposal tile into LDS (wave 0 only) ---
  if (tid < 32) {
    tdm_load_tile(&box[0], props + (size_t)bl * NMS_PRE, NMS_PRE * 16);
    __builtin_amdgcn_s_wait_tensorcnt(0);
  }
  __syncthreads();
  // Portable guarantee fill (TDM descriptor packing not HW-verified) + scores.
  for (int i = tid; i < NMS_PRE; i += NMS_T) {
    box[i]  = props[(size_t)bl * NMS_PRE + i];
    sc[i]   = topS[(size_t)bl * NMS_PRE + i];
    supp[i] = 0;
  }
  __syncthreads();

  // --- greedy NMS over score-sorted boxes ---
  for (int i = 0; i < NMS_PRE; ++i) {
    if (!supp[i]) {
      float4 bi4 = box[i];
      float ai = (bi4.z - bi4.x) * (bi4.w - bi4.y);
      for (int j = i + 1 + tid; j < NMS_PRE; j += NMS_T) {
        if (supp[j]) continue;
        float4 bj = box[j];
        float xx1 = fmaxf(bi4.x, bj.x), yy1 = fmaxf(bi4.y, bj.y);
        float xx2 = fminf(bi4.z, bj.z), yy2 = fminf(bi4.w, bj.w);
        float iw = fmaxf(xx2 - xx1, 0.f), ih = fmaxf(yy2 - yy1, 0.f);
        float inter = iw * ih;
        float aj = (bj.z - bj.x) * (bj.w - bj.y);
        float uni = fmaxf(ai + aj - inter, 1e-6f);
        if (inter > 0.7f * uni) supp[j] = 1;
      }
    }
    __syncthreads();
  }

  // keep flag: survives NMS AND is a real score (> -1) -> ranks before the -1 ties
  for (int i = tid; i < 2048; i += NMS_T) {
    int f = 0;
    if (i < NMS_PRE) f = (!supp[i] && sc[i] > -1.0f) ? 1 : 0;
    kf[i] = (_Float16)f;
  }
  __syncthreads();

  // --- WMMA prefix scan: 8 waves x 256 elems, D1 = L*X ; D2 = T*U + D1 ---
  {
    const int lane = tid & 31;
    const int w    = tid >> 5;       // chunk id, 0..7
    const int n    = lane & 15;      // A-row m / B-col n / D-col n
    v16h A;
    #pragma unroll
    for (int h = 0; h < 16; ++h) {   // 16-bit A 16x32, contiguous k = 16*(lane/16) + h
      int p = h >> 1, sb = h & 1;
      int K = ((lane < 16) ? 0 : 16) + h;
      A[h] = (_Float16)((K < 16 && K <= n) ? 1.0f : 0.0f);   // lower-tri incl diag
    }
    v16h Bm = {};                     // B 32x16: lane n holds K=0..15, lane n+16 K=16..31
    if (lane < 16) Bm = *(const v16h*)&kf[w * 256 + lane * 16];
    v8f c0 = {};
    v8f d1 = __builtin_amdgcn_wmma_f32_16x16x32_f16(false, A, false, Bm,
                                                    (short)0, c0, false, false);
    // segment totals T[k] = D1[15,k] live in lane k+16, reg 7
    v16h A2;
    #pragma unroll
    for (int h = 0; h < 16; ++h) {
      int p = h >> 1, sb = h & 1;
      int K = ((lane < 16) ? 0 : 16) + h;
      float tv = __shfl(d1[7], (K & 15) + 16, 32);
      A2[h] = (_Float16)((K < 16) ? tv : 0.0f);
    }
    v16h B2 = {};
    if (lane < 16) {
      #pragma unroll
      for (int h = 0; h < 16; ++h) B2[h] = (_Float16)((h < n) ? 1.0f : 0.0f); // strict upper
    }
    v8f d2 = __builtin_amdgcn_wmma_f32_16x16x32_f16(false, A2, false, B2,
                                                    (short)0, d1, false, false);
    float wtot = __shfl(d2[7], 31, 32);          // D2[15,15] = chunk total
    if (lane == 0) ctot[w] = wtot;
    __syncthreads();
    float off = 0.f;
    for (int p = 0; p < 8; ++p) if (p < w) off += ctot[p];
    #pragma unroll
    for (int r = 0; r < 8; ++r) {
      int m = r + ((lane < 16) ? 0 : 8);
      scanv[w * 256 + n * 16 + m] = (int)(d2[r] + off + 0.5f);  // global inclusive
    }
  }
  __syncthreads();

  const int C1 = scanv[NMS_PRE - 1];   // total kept-with-real-score
  for (int i = tid; i < NMS_PRE; i += NMS_T) {
    int f    = (!supp[i] && sc[i] > -1.0f) ? 1 : 0;
    int incl = scanv[i];
    int excl = incl - f;
    int pos  = f ? excl : (C1 + (i - excl));     // kept first, then -1 ties by index
    if (pos < NMS_POST) {
      lvlS[(size_t)bl * NMS_POST + pos] = f ? sc[i] : -1.0f;
      lvlB[(size_t)bl * NMS_POST + pos] = box[i];
    }
  }
}

// ---------------------------------------------------------------------------
// Stage 4: exact cross-level top-1000 via rank-by-binary-search merge
// ---------------------------------------------------------------------------
__global__ void final_kernel(const float* __restrict__ lvlS,
                             const float4* __restrict__ lvlB,
                             float4* __restrict__ out) {
  int t = blockIdx.x * blockDim.x + threadIdx.x;
  if (t >= 2 * 5 * NMS_POST) return;
  int b   = t / (5 * NMS_POST);
  int g   = t % (5 * NMS_POST);       // global index within concat (tie-break key)
  int lvl = g / NMS_POST, pos = g % NMS_POST;
  const float* S = lvlS + (size_t)b * 5 * NMS_POST;
  float s = S[g];
  int rank = pos;                      // elements of own (sorted) list that beat us
  for (int L = 0; L < 5; ++L) {
    if (L == lvl) continue;
    const float* SL = S + (size_t)L * NMS_POST;
    int gbase = L * NMS_POST;
    int lo = 0, hi = NMS_POST;
    while (lo < hi) {
      int mid = (lo + hi) >> 1;
      float sv = SL[mid];
      bool bt = (sv > s) || ((sv == s) && (gbase + mid < g));
      if (bt) lo = mid + 1; else hi = mid;
    }
    rank += lo;
  }
  if (rank < MAX_NUM)
    out[(size_t)b * MAX_NUM + rank] = lvlB[(size_t)b * 5 * NMS_POST + g];
}

// ---------------------------------------------------------------------------
extern "C" void kernel_launch(void* const* d_in, const int* in_sizes, int n_in,
                              void* d_out, int out_size, void* d_ws, size_t ws_size,
                              hipStream_t stream) {
  (void)in_sizes; (void)n_in; (void)out_size; (void)ws_size;
  const float *cls[5], *bbx[5], *anc[5];
  for (int l = 0; l < 5; ++l) {
    cls[l] = (const float*)d_in[3 * l + 0];
    bbx[l] = (const float*)d_in[3 * l + 1];
    anc[l] = (const float*)d_in[3 * l + 2];
  }
  // Workspace layout (all 16B-aligned): 680 KB total
  float*  top_s = (float*)d_ws;                       // 10*2000 f32
  int*    top_i = (int*)(top_s + 10 * NMS_PRE);       // 10*2000 i32
  float4* props = (float4*)(top_i + 10 * NMS_PRE);    // 10*2000 float4
  float*  lvl_s = (float*)(props + 10 * NMS_PRE);     // 10*1000 f32
  float4* lvl_b = (float4*)(lvl_s + 10 * NMS_POST);   // 10*1000 float4

  topk_kernel<<<10, TK_T, 0, stream>>>(cls[0], cls[1], cls[2], cls[3], cls[4],
                                       top_s, top_i);
  decode_kernel<<<(2 * 5 * NMS_PRE + 255) / 256, 256, 0, stream>>>(
      bbx[0], bbx[1], bbx[2], bbx[3], bbx[4],
      anc[0], anc[1], anc[2], anc[3], anc[4], top_i, props);
  nms_kernel<<<10, NMS_T, 0, stream>>>(props, top_s, lvl_s, lvl_b);
  final_kernel<<<(2 * 5 * NMS_POST + 255) / 256, 256, 0, stream>>>(
      lvl_s, lvl_b, (float4*)d_out);
}
